// SolvGNNV3_37778532335672
// MI455X (gfx1250) — matrix operand order repaired and checked by reading, not compiled, not run on hardware
//
#include <hip/hip_runtime.h>

// ---- problem constants (match reference) ----
#define N_NODES  100000   // multiple of 16
#define N_EDGES  800000
#define N_GRAPHS 2048     // multiple of 64
#define DIM_IN   74
#define K_IN_PAD 96       // DIM_IN padded to multiple of 32
#define DIM_H    256
#define DIM_M1   1024
#define DIM_M2   512

typedef __attribute__((ext_vector_type(16))) __bf16 v16bf;
typedef __attribute__((ext_vector_type(8)))  float  v8f;

__device__ __forceinline__ void atomic_add_f32(float* p, float v) {
  __hip_atomic_fetch_add(p, v, __ATOMIC_RELAXED, __HIP_MEMORY_SCOPE_AGENT);
}

// ---------------- utility kernels ----------------
__global__ void k_zero(float* p, long long n) {
  long long i = (long long)blockIdx.x * blockDim.x + threadIdx.x;
  long long s = (long long)gridDim.x * blockDim.x;
  for (; i < n; i += s) p[i] = 0.0f;
}

__global__ void k_degree(const int* __restrict__ src, const int* __restrict__ dst,
                         float* outdeg, float* indeg, int E) {
  int e = blockIdx.x * blockDim.x + threadIdx.x;
  if (e < E) {
    atomic_add_f32(&outdeg[src[e]], 1.0f);
    atomic_add_f32(&indeg[dst[e]], 1.0f);
  }
}

__global__ void k_norm(float* a, float* b, int n) {
  int i = blockIdx.x * blockDim.x + threadIdx.x;
  if (i < n) {
    a[i] = rsqrtf(fmaxf(a[i], 1.0f));
    b[i] = rsqrtf(fmaxf(b[i], 1.0f));
  }
}

// Pack fp32 W[Kact x Nact] (row-major) into bf16 WMMA-B fragment order:
// Wpack[ntile][ktile][lane][i], i=0..15 contiguous per lane (32B per lane).
// B 32x16 lane layout: elems 0..7 -> K=k0+8h+i, elems 8..15 -> K=k0+16+8h+(i-8) (h=lane>>4); N = lane&15.
__global__ void k_wpack(const float* __restrict__ W, __bf16* __restrict__ Wpack,
                        int Kact, int Nact, int Kpad, int Npad) {
  int nKt = Kpad >> 5;
  long long total = (long long)(Npad >> 4) * nKt * 512;
  long long idx = (long long)blockIdx.x * blockDim.x + threadIdx.x;
  if (idx >= total) return;
  int i    = idx & 15;
  int lane = (idx >> 4) & 31;
  long long rest = idx >> 9;
  int kt = (int)(rest % nKt);
  int nt = (int)(rest / nKt);
  int K   = kt * 32 + ((lane >> 4) << 3) + (i & 7) + ((i >> 3) << 4);
  int col = nt * 16 + (lane & 15);
  float v = (K < Kact && col < Nact) ? W[(long long)K * Nact + col] : 0.0f;
  Wpack[idx] = (__bf16)v;
}

// m[dst*mstride + t] += x[src*width + t] * out_norm[src]
__global__ void k_spmm(const float* __restrict__ x, const float* __restrict__ outnorm,
                       const int* __restrict__ src, const int* __restrict__ dst,
                       float* __restrict__ m, int width, int mstride) {
  int e = blockIdx.x;
  int t = threadIdx.x;
  if (t >= width) return;
  int s = src[e], d = dst[e];
  float v = x[(long long)s * width + t] * outnorm[s];
  atomic_add_f32(&m[(long long)d * mstride + t], v);
}

// ---------------- WMMA GEMM ----------------
// C[M x Npad] = act( (A[M x KPAD] * rowScale) @ Wpack + bias )
// A: fp32, row stride = KPAD floats, zero-padded in K (no bounds checks).
// M must be a multiple of 16. One wave -> 16 x (16*NT) output (NT accumulators,
// A fragment loaded/converted once and reused NT times). 4 waves/block along M.
// ACT: 0=none, 1=relu, 2=leaky_relu(0.01)
template <int KPAD, int NT, int ACT>
__global__ void k_gemm_wmma(const float* __restrict__ A, const float* __restrict__ rowScale,
                            const __bf16* __restrict__ Wpack, const float* __restrict__ bias,
                            float* __restrict__ C, int M, int Npad, int Nact) {
  constexpr int NKT = KPAD / 32;
  const int lane = threadIdx.x & 31;
  const int wave = threadIdx.x >> 5;
  const int nt0 = blockIdx.x * NT;
  const int m0 = (blockIdx.y * 4 + wave) * 16;
  if (m0 >= M) return;                           // whole-wave skip (M % 16 == 0)

  const int mrow  = m0 + (lane & 15);
  const int koffA = (lane >> 4) << 3;            // A: lanes 0-15 -> +0, 16-31 -> +8

  const float rs = rowScale ? rowScale[mrow] : 1.0f;
  const float*  Arow = A + (long long)mrow * KPAD;
  const __bf16* Wb   = Wpack + (((long long)nt0 * NKT * 32 + lane) << 4);

  v8f acc[NT];
#pragma unroll
  for (int j = 0; j < NT; ++j) acc[j] = (v8f){};

#pragma unroll
  for (int kt = 0; kt < NKT; ++kt) {
    const int kaA = kt * 32 + koffA;
    // A 16x32 layout: elems 0..7 -> K=kaA+i ; elems 8..15 -> K=kaA+16+i
    const float4 p0 = *(const float4*)(Arow + kaA);
    const float4 p1 = *(const float4*)(Arow + kaA + 4);
    const float4 p2 = *(const float4*)(Arow + kaA + 16);
    const float4 p3 = *(const float4*)(Arow + kaA + 20);
    v16bf a;
    a[0]  = (__bf16)(p0.x * rs); a[1]  = (__bf16)(p0.y * rs);
    a[2]  = (__bf16)(p0.z * rs); a[3]  = (__bf16)(p0.w * rs);
    a[4]  = (__bf16)(p1.x * rs); a[5]  = (__bf16)(p1.y * rs);
    a[6]  = (__bf16)(p1.z * rs); a[7]  = (__bf16)(p1.w * rs);
    a[8]  = (__bf16)(p2.x * rs); a[9]  = (__bf16)(p2.y * rs);
    a[10] = (__bf16)(p2.z * rs); a[11] = (__bf16)(p2.w * rs);
    a[12] = (__bf16)(p3.x * rs); a[13] = (__bf16)(p3.y * rs);
    a[14] = (__bf16)(p3.z * rs); a[15] = (__bf16)(p3.w * rs);
#pragma unroll
    for (int j = 0; j < NT; ++j) {
      // B fragment for sub-tile j: 32 contiguous bytes per lane (prepacked)
      const v16bf b = *(const v16bf*)(Wb + (((long long)j * NKT + kt) << 9));
      acc[j] = __builtin_amdgcn_wmma_f32_16x16x32_bf16(
          false, a, false, b, (short)0, acc[j], false, false);
    }
  }

#pragma unroll
  for (int j = 0; j < NT; ++j) {
    const int col = (nt0 + j) * 16 + (lane & 15);
    const float bv = (col < Nact) ? bias[col] : 0.0f;
#pragma unroll
    for (int v = 0; v < 8; ++v) {
      int mr = m0 + v + ((lane >> 4) << 3);      // C layout: lanes 16-31 -> M+8
      float val = acc[j][v] + bv;
      if (ACT == 1) val = fmaxf(val, 0.0f);
      if (ACT == 2) val = (val > 0.0f) ? val : 0.01f * val;
      C[(long long)mr * Npad + col] = val;
    }
  }
}

// ---------------- pooling / output ----------------
__global__ void k_pool(const float* __restrict__ x, const int* __restrict__ gid,
                       float* sums, float* cnts) {
  int n = blockIdx.x;
  int t = threadIdx.x;
  int g = gid[n];
  atomic_add_f32(&sums[(long long)g * DIM_H + t], x[(long long)n * DIM_H + t]);
  if (t == 0) atomic_add_f32(&cnts[g], 1.0f);
}

__global__ void k_mean(float* sums, const float* __restrict__ cnts) {
  int i = blockIdx.x * blockDim.x + threadIdx.x;
  if (i < N_GRAPHS * DIM_H) {
    int g = i / DIM_H;
    sums[i] /= fmaxf(cnts[g], 1.0f);
  }
}

__global__ void k_extract(const float* __restrict__ z3, float* __restrict__ out) {
  int g = blockIdx.x * blockDim.x + threadIdx.x;
  if (g < N_GRAPHS) out[g] = z3[(long long)g * 16];
}

// ---------------- host launcher ----------------
extern "C" void kernel_launch(void* const* d_in, const int* in_sizes, int n_in,
                              void* d_out, int out_size, void* d_ws, size_t ws_size,
                              hipStream_t stream) {
  (void)in_sizes; (void)n_in; (void)out_size; (void)ws_size;
  const float* h     = (const float*)d_in[0];
  const int*   src   = (const int*)d_in[1];
  const int*   dst   = (const int*)d_in[2];
  const int*   gid   = (const int*)d_in[3];
  const float* W_in  = (const float*)d_in[4];
  const float* b_in  = (const float*)d_in[5];
  const float* W_gcr = (const float*)d_in[6];
  const float* b_gcr = (const float*)d_in[7];
  const float* W1    = (const float*)d_in[8];
  const float* b1    = (const float*)d_in[9];
  const float* W2    = (const float*)d_in[10];
  const float* b2    = (const float*)d_in[11];
  const float* W3    = (const float*)d_in[12];
  const float* b3    = (const float*)d_in[13];
  float* out = (float*)d_out;

  char* ws = (char*)d_ws;
  size_t off = 0;
  auto alloc = [&](size_t bytes) -> void* {
    void* p = ws + off;
    off += (bytes + 255) & ~(size_t)255;
    return p;
  };

  float*  outnorm = (float*)alloc((size_t)N_NODES * 4);
  float*  innorm  = (float*)alloc((size_t)N_NODES * 4);
  float*  bufX    = (float*)alloc((size_t)N_NODES * DIM_H * 4);
  float*  bufM    = (float*)alloc((size_t)N_NODES * DIM_H * 4);   // also used N x 96 in layer0
  __bf16* Wp_in   = (__bf16*)alloc((size_t)K_IN_PAD * 256 * 2);
  __bf16* Wp_gcr  = (__bf16*)alloc((size_t)10 * 256 * 256 * 2);
  __bf16* Wp1     = (__bf16*)alloc((size_t)256 * 1024 * 2);
  __bf16* Wp2     = (__bf16*)alloc((size_t)1024 * 512 * 2);
  __bf16* Wp3     = (__bf16*)alloc((size_t)512 * 16 * 2);
  float*  sums    = (float*)alloc((size_t)N_GRAPHS * DIM_H * 4);
  float*  cnts    = (float*)alloc((size_t)N_GRAPHS * 4);
  float*  z1      = (float*)alloc((size_t)N_GRAPHS * DIM_M1 * 4);
  float*  z2      = (float*)alloc((size_t)N_GRAPHS * DIM_M2 * 4);
  float*  z3      = (float*)alloc((size_t)N_GRAPHS * 16 * 4);

  // degrees -> norms
  k_zero<<<1024, 256, 0, stream>>>(outnorm, N_NODES);
  k_zero<<<1024, 256, 0, stream>>>(innorm, N_NODES);
  k_degree<<<(N_EDGES + 255) / 256, 256, 0, stream>>>(src, dst, outnorm, innorm, N_EDGES);
  k_norm<<<(N_NODES + 255) / 256, 256, 0, stream>>>(outnorm, innorm, N_NODES);

  // weights -> packed bf16 fragments
  auto packLaunch = [&](const float* W, __bf16* Wp, int Kact, int Nact, int Kpad, int Npad) {
    long long tot = (long long)(Npad / 16) * (Kpad / 32) * 512;
    k_wpack<<<(unsigned)((tot + 255) / 256), 256, 0, stream>>>(W, Wp, Kact, Nact, Kpad, Npad);
  };
  packLaunch(W_in, Wp_in, DIM_IN, DIM_H, K_IN_PAD, 256);
  for (int i = 0; i < 10; ++i)
    packLaunch(W_gcr + (size_t)i * DIM_H * DIM_H, Wp_gcr + (size_t)i * 256 * 256, 256, 256, 256, 256);
  packLaunch(W1, Wp1, 256, 1024, 256, 1024);
  packLaunch(W2, Wp2, 1024, 512, 1024, 512);
  packLaunch(W3, Wp3, 512, 1, 512, 16);

  const dim3 gridConv(DIM_H / 64, (N_NODES / 16 + 3) / 4);  // NT=4 -> 64 cols per wave

  // input conv: h (N x 74) -> bufM (N x 96, zero padded) -> bufX (N x 256)
  k_zero<<<2048, 256, 0, stream>>>(bufM, (long long)N_NODES * K_IN_PAD);
  k_spmm<<<N_EDGES, 128, 0, stream>>>(h, outnorm, src, dst, bufM, DIM_IN, K_IN_PAD);
  k_gemm_wmma<K_IN_PAD, 4, 0><<<gridConv, 128, 0, stream>>>(bufM, innorm, Wp_in, b_in, bufX,
                                                            N_NODES, DIM_H, DIM_H);

  // 5 GCR blocks: conv->relu->conv->relu (relu fused into GEMM epilogues)
  for (int i = 0; i < 5; ++i) {
    for (int j = 0; j < 2; ++j) {
      int w = 2 * i + j;
      k_zero<<<4096, 256, 0, stream>>>(bufM, (long long)N_NODES * DIM_H);
      k_spmm<<<N_EDGES, 256, 0, stream>>>(bufX, outnorm, src, dst, bufM, DIM_H, DIM_H);
      k_gemm_wmma<DIM_H, 4, 1><<<gridConv, 128, 0, stream>>>(bufM, innorm,
                                                             Wp_gcr + (size_t)w * 256 * 256,
                                                             b_gcr + (size_t)w * DIM_H, bufX,
                                                             N_NODES, DIM_H, DIM_H);
    }
  }

  // mean pooling per graph
  k_zero<<<1024, 256, 0, stream>>>(sums, (long long)N_GRAPHS * DIM_H);
  k_zero<<<8, 256, 0, stream>>>(cnts, N_GRAPHS);
  k_pool<<<N_NODES, DIM_H, 0, stream>>>(bufX, gid, sums, cnts);
  k_mean<<<(N_GRAPHS * DIM_H + 255) / 256, 256, 0, stream>>>(sums, cnts);

  // MLP head
  dim3 g1(DIM_M1 / 64, N_GRAPHS / 64);
  k_gemm_wmma<DIM_H, 4, 2><<<g1, 128, 0, stream>>>(sums, nullptr, Wp1, b1, z1,
                                                   N_GRAPHS, DIM_M1, DIM_M1);
  dim3 g2(DIM_M2 / 64, N_GRAPHS / 64);
  k_gemm_wmma<DIM_M1, 4, 2><<<g2, 128, 0, stream>>>(z1, nullptr, Wp2, b2, z2,
                                                    N_GRAPHS, DIM_M2, DIM_M2);
  dim3 g3(1, N_GRAPHS / 64);
  k_gemm_wmma<DIM_M2, 1, 0><<<g3, 128, 0, stream>>>(z2, nullptr, Wp3, b3, z3,
                                                    N_GRAPHS, 16, 1);
  k_extract<<<(N_GRAPHS + 255) / 256, 256, 0, stream>>>(z3, out);
}
